// TransformerBlock_8349416424041
// MI455X (gfx1250) — matrix operand order repaired and checked by reading, not compiled, not run on hardware
//
#include <hip/hip_runtime.h>

typedef __attribute__((ext_vector_type(16))) _Float16 v16h;
typedef __attribute__((ext_vector_type(8)))  _Float16 v8h;
typedef __attribute__((ext_vector_type(2)))  _Float16 f16x2;
typedef __attribute__((ext_vector_type(8)))  float    v8f;
typedef __attribute__((ext_vector_type(4)))  int      v4i;

constexpr int kB  = 8;
constexpr int kS  = 1024;
constexpr int kD  = 1024;
constexpr int kH  = 16;
constexpr int kHD = 64;
constexpr int kM  = kB * kS;   // 8192 flattened rows

#define AS1 __attribute__((address_space(1)))
#define AS3 __attribute__((address_space(3)))

#if __has_builtin(__builtin_amdgcn_global_load_async_to_lds_b128) && \
    __has_builtin(__builtin_amdgcn_s_wait_asynccnt)
#define USE_ASYNC_LDS 1
#else
#define USE_ASYNC_LDS 0
#endif

// 16B global -> LDS copy: async (ASYNCcnt-tracked) when available
__device__ __forceinline__ void copy16_g2l(_Float16* l, const _Float16* g) {
#if USE_ASYNC_LDS
  v4i* gsrc = (v4i*)const_cast<_Float16*>(g);
  v4i* ldst = (v4i*)l;
  __builtin_amdgcn_global_load_async_to_lds_b128((AS1 v4i*)gsrc, (AS3 v4i*)ldst,
                                                 0, 0);
#else
  *(v8h*)l = *(const v8h*)g;
#endif
}

template <int N>
__device__ __forceinline__ void wait_async() {
#if USE_ASYNC_LDS
  __builtin_amdgcn_s_wait_asynccnt(N);
#endif
}

// ---------------------------------------------------------------------------
// WMMA fragment layouts (CDNA5 16-bit, wave32):
//  A (16x32, MxK): lane L holds row M=L&15; K = (L>>4)*16 + {0..15} (this code's contiguous k order, not AMD's table order).
//  B (32x16, KxN): lane L holds col N=L&15; K = (L>>4)*16 + {0..15} (same contiguous k order as A, not AMD's table order).
//  C/D (16x16 f32): lane L holds col N=L&15; VGPR r -> row M=(L>>4)*8+r.
// ---------------------------------------------------------------------------
__device__ __forceinline__ v16h load_afrag(const _Float16* base, int lda,
                                           int row0, int k0, int qr, int hs) {
  const _Float16* p = base + (size_t)(row0 + qr) * (size_t)lda + k0 + hs * 16;
  v16h a;
  *(v8h*)&a       = *(const v8h*)p;
  *((v8h*)&a + 1) = *(const v8h*)(p + 8);
  return a;
}

__device__ __forceinline__ v16h load_bfrag(const _Float16* base, int ldb,
                                           int col0, int k0, int qr, int hs) {
  const _Float16* p = base + (size_t)(col0 + qr) * (size_t)ldb + k0 + hs * 16;
  v16h b;
  *(v8h*)&b       = *(const v8h*)p;
  *((v8h*)&b + 1) = *(const v8h*)(p + 8);
  return b;
}

__device__ __forceinline__ v8f wmma_fp16(v16h a, v16h b, v8f c) {
  return __builtin_amdgcn_wmma_f32_16x16x32_f16(false, a, false, b, (short)0, c,
                                                false, false);
}

// ---------------------------------------------------------------------------
// LayerNorm (fp32 in) -> f16 activations, one block per row of 1024
// ---------------------------------------------------------------------------
__global__ __launch_bounds__(256) void ln_f16_kernel(const float* __restrict__ x,
                                                     const float* __restrict__ g,
                                                     const float* __restrict__ b,
                                                     _Float16* __restrict__ out) {
  const int row = blockIdx.x;
  const float* xr = x + (size_t)row * kD;
  float s = 0.f, s2 = 0.f;
  for (int c = threadIdx.x; c < kD; c += 256) {
    float v = xr[c];
    s += v;
    s2 += v * v;
  }
  for (int m = 16; m >= 1; m >>= 1) {
    s  += __shfl_xor(s, m, 32);
    s2 += __shfl_xor(s2, m, 32);
  }
  __shared__ float sh[16];
  const int w = threadIdx.x >> 5, lane = threadIdx.x & 31;
  if (lane == 0) { sh[w] = s; sh[8 + w] = s2; }
  __syncthreads();
  s = 0.f; s2 = 0.f;
  for (int i = 0; i < 8; ++i) { s += sh[i]; s2 += sh[8 + i]; }
  const float mu  = s * (1.f / kD);
  const float var = s2 * (1.f / kD) - mu * mu;
  const float rs  = rsqrtf(var + 1e-5f);
  _Float16* orow = out + (size_t)row * kD;
  for (int c = threadIdx.x; c < kD; c += 256)
    orow[c] = (_Float16)((xr[c] - mu) * rs * g[c] + b[c]);
}

// ---------------------------------------------------------------------------
// Weight convert + transpose: f32 W[K][N] -> f16 Wt[N][K]
// grid = (N/256, K): no integer division, coalesced reads
// ---------------------------------------------------------------------------
__global__ __launch_bounds__(256) void wtrans_kernel(const float* __restrict__ W,
                                                     _Float16* __restrict__ Wt,
                                                     int K, int N) {
  const int n = blockIdx.x * 256 + threadIdx.x;
  const int k = blockIdx.y;
  Wt[(size_t)n * K + k] = (_Float16)W[(size_t)k * N + n];
}

// ---------------------------------------------------------------------------
// f16 GEMM:  C[M][N] = A[M][K] @ Bt[N][K]^T  + epilogue
// block = 256 thr = 8 waves (4x2), wave tile 32x64, block tile 128x128.
// K-slabs of 32 double-buffered in LDS via async global->LDS loads; fragments
// read back with ds_load_b128 through a chunk-XOR swizzle (conflict-free).
// ---------------------------------------------------------------------------
struct GemmArgs {
  const _Float16* A;      // [M][K] f16 row-major
  const _Float16* Bt;     // [N][K] f16 row-major (i.e. B transposed)
  const float*    bias;   // [N]
  const float*    resid;  // [M][N] f32 (EPI_F32_RESID)
  float*          outF;   // f32 out
  _Float16*       outH;   // f16 out (EPI_F16_GELU)
  _Float16*       q;      // EPI_QKV: [B*H][S][64]   (pre-scaled by 1/sqrt(HD))
  _Float16*       k;      // EPI_QKV: [B*H][S][64]
  _Float16*       vt;     // EPI_QKV: [B*H][64][S]   (V transposed)
  int M, N, K;
};

enum { EPI_QKV = 0, EPI_F32_RESID = 1, EPI_F16_GELU = 2 };

// swizzled half-offset of 16B chunk c (0..3) in a 128x32 f16 LDS slab
__device__ __forceinline__ int swz(int row, int c) {
  return row * 32 + ((c ^ ((row >> 2) & 3)) * 8);
}

template <int EPI>
__global__ __launch_bounds__(256) void gemm_f16_kernel(GemmArgs p) {
  __shared__ _Float16 lsA[2][128 * 32];   // 8 KB per buffer
  __shared__ _Float16 lsB[2][128 * 32];   // 32 KB total of 320 KB WGP LDS
  const int tid  = threadIdx.x;
  const int lane = tid & 31, w = tid >> 5;
  const int wr = w >> 1, wc = w & 1;
  const int mBlk = blockIdx.y * 128, nBlk = blockIdx.x * 128;
  const int m0 = wr * 32, n0 = wc * 64;           // LDS-local origins
  const int qr = lane & 15, hs = lane >> 4;

  // stage one 128x32 A-slab + 128x32 B-slab (4 x 16B per thread)
  auto stage = [&](int buf, int k0) {
    for (int rep = 0; rep < 2; ++rep) {
      const int cid = tid + rep * 256;            // 0..511 chunk id
      const int row = cid >> 2, c = cid & 3;
      const int lo = swz(row, c);
      copy16_g2l(&lsA[buf][lo], p.A  + (size_t)(mBlk + row) * p.K + k0 + c * 8);
      copy16_g2l(&lsB[buf][lo], p.Bt + (size_t)(nBlk + row) * p.K + k0 + c * 8);
    }
  };
  auto afragL = [&](const _Float16* base, int row0) -> v16h {
    const int row = row0 + qr;
    v16h a;
    *(v8h*)&a       = *(const v8h*)(base + swz(row, hs * 2));
    *((v8h*)&a + 1) = *(const v8h*)(base + swz(row, hs * 2 + 1));
    return a;
  };
  auto bfragL = [&](const _Float16* base, int col0) -> v16h {
    const int row = col0 + qr;
    v16h b;
    *(v8h*)&b       = *(const v8h*)(base + swz(row, hs * 2));
    *((v8h*)&b + 1) = *(const v8h*)(base + swz(row, hs * 2 + 1));
    return b;
  };

  v8f acc[2][4] = {};
  const int nsteps = p.K / 32;
  stage(0, 0);
  for (int s = 0; s < nsteps; ++s) {
    const int buf = s & 1;
    if (s + 1 < nsteps) {
      stage(buf ^ 1, (s + 1) * 32);   // issue next slab (async)
      wait_async<4>();                // slab s complete (in-order dones)
    } else {
      wait_async<0>();
    }
    __syncthreads();
    v16h a0 = afragL(lsA[buf], m0);
    v16h a1 = afragL(lsA[buf], m0 + 16);
    v16h b0 = bfragL(lsB[buf], n0);
    v16h b1 = bfragL(lsB[buf], n0 + 16);
    v16h b2 = bfragL(lsB[buf], n0 + 32);
    v16h b3 = bfragL(lsB[buf], n0 + 48);
    acc[0][0] = wmma_fp16(a0, b0, acc[0][0]);
    acc[0][1] = wmma_fp16(a0, b1, acc[0][1]);
    acc[0][2] = wmma_fp16(a0, b2, acc[0][2]);
    acc[0][3] = wmma_fp16(a0, b3, acc[0][3]);
    acc[1][0] = wmma_fp16(a1, b0, acc[1][0]);
    acc[1][1] = wmma_fp16(a1, b1, acc[1][1]);
    acc[1][2] = wmma_fp16(a1, b2, acc[1][2]);
    acc[1][3] = wmma_fp16(a1, b3, acc[1][3]);
    __syncthreads();                  // done reading buf before it is rewritten
  }

  for (int i = 0; i < 2; ++i) {
    for (int j = 0; j < 4; ++j) {
      const int col = nBlk + n0 + j * 16 + qr;
      const float bv = p.bias[col];
      for (int rr = 0; rr < 8; ++rr) {
        const int row = mBlk + m0 + i * 16 + hs * 8 + rr;
        const float v = acc[i][j][rr] + bv;
        if (EPI == EPI_F32_RESID) {
          const size_t idx = (size_t)row * p.N + col;
          p.outF[idx] = v + p.resid[idx];
        } else if (EPI == EPI_F16_GELU) {
          const float gv = 0.5f * v * (1.f + erff(v * 0.70710678118654752f));
          p.outH[(size_t)row * p.N + col] = (_Float16)gv;
        } else {  // EPI_QKV: col in [0,3D) = (part, head, hd)
          const int part = col >> 10;
          const int rem  = col & 1023;
          const int hh = rem >> 6, hd = rem & 63;
          const int bb = row >> 10, ss = row & 1023;
          const int bh = bb * kH + hh;
          if (part == 0) {  // fold 1/sqrt(HD) into Q
            p.q[((size_t)bh * kS + ss) * kHD + hd] = (_Float16)(v * 0.125f);
          } else if (part == 1) {
            p.k[((size_t)bh * kS + ss) * kHD + hd] = (_Float16)v;
          } else {
            p.vt[((size_t)bh * kHD + hd) * kS + ss] = (_Float16)v;
          }
        }
      }
    }
  }
}

// ---------------------------------------------------------------------------
// Flash attention (causal, per-(b,h)); wave handles a 16-row q tile.
// S^T = K_blk (A) x q^T (B)  and  O^T += V^T (A) x P^T (B).
// Software-pipelined: next K-fragments prefetched before current WMMAs;
// V-fragments loaded before the softmax VALU work to hide latency.
// ---------------------------------------------------------------------------
__global__ __launch_bounds__(256) void flash_attn_kernel(
    const _Float16* __restrict__ qb, const _Float16* __restrict__ kb,
    const _Float16* __restrict__ vtb, _Float16* __restrict__ ctx) {
  const int lane = threadIdx.x & 31;
  const int w    = threadIdx.x >> 5;
  const int qr = lane & 15, hs = lane >> 4;
  const int bh = blockIdx.y;
  const int bIdx = bh >> 4, hIdx = bh & 15;
  const int q0 = blockIdx.x * 128 + w * 16;

  const _Float16* Q  = qb  + (size_t)bh * kS * kHD;
  const _Float16* Km = kb  + (size_t)bh * kS * kHD;
  const _Float16* Vt = vtb + (size_t)bh * kHD * kS;

  // q^T B-fragments (N = q row, K = head dim), hoisted out of the key loop
  const v16h bq0 = load_bfrag(Q, kHD, q0, 0,  qr, hs);
  const v16h bq1 = load_bfrag(Q, kHD, q0, 32, qr, hs);

  float mrow = -__builtin_inff();
  float lrow = 0.f;
  v8f o[4] = {};

  const int qg = q0 + qr;       // this lane's global query row
  const int kend = q0 + 16;     // causal: keys <= last row of tile

  v16h ak[4];                   // current K fragments (prefetched)
  ak[0] = load_afrag(Km, kHD, 0,  0,  qr, hs);
  ak[1] = load_afrag(Km, kHD, 0,  32, qr, hs);
  ak[2] = load_afrag(Km, kHD, 16, 0,  qr, hs);
  ak[3] = load_afrag(Km, kHD, 16, 32, qr, hs);

  for (int kk = 0; kk < kend; kk += 32) {
    const bool more = (kk + 32) < kend;   // wave-uniform
    // issue loads early: V for this step, K for the next step
    v16h av[4];
    for (int dt = 0; dt < 4; ++dt)
      av[dt] = load_afrag(Vt, kS, dt * 16, kk, qr, hs);
    v16h nak[4];
    if (more) {
      nak[0] = load_afrag(Km, kHD, kk + 32, 0,  qr, hs);
      nak[1] = load_afrag(Km, kHD, kk + 32, 32, qr, hs);
      nak[2] = load_afrag(Km, kHD, kk + 48, 0,  qr, hs);
      nak[3] = load_afrag(Km, kHD, kk + 48, 32, qr, hs);
    }

    v8f st0 = {}, st1 = {};               // S^T tiles: rows=key, cols=q
    st0 = wmma_fp16(ak[0], bq0, st0);
    st0 = wmma_fp16(ak[1], bq1, st0);
    st1 = wmma_fp16(ak[2], bq0, st1);
    st1 = wmma_fp16(ak[3], bq1, st1);

    float s0[8], s1[8];
    float vmax = -__builtin_inff();
    for (int rr = 0; rr < 8; ++rr) {
      const int key0 = kk + hs * 8 + rr;
      const int key1 = key0 + 16;
      s0[rr] = (key0 <= qg) ? st0[rr] : -__builtin_inff();  // Q pre-scaled
      s1[rr] = (key1 <= qg) ? st1[rr] : -__builtin_inff();
      vmax = fmaxf(vmax, fmaxf(s0[rr], s1[rr]));
    }
    vmax = fmaxf(vmax, __shfl_xor(vmax, 16, 32));
    const float mnew  = fmaxf(mrow, vmax);
    const float alpha = expf(mrow - mnew);
    float rsum = 0.f;
    for (int rr = 0; rr < 8; ++rr) {
      s0[rr] = expf(s0[rr] - mnew);
      s1[rr] = expf(s1[rr] - mnew);
      rsum += s0[rr] + s1[rr];
    }
    rsum += __shfl_xor(rsum, 16, 32);
    lrow = lrow * alpha + rsum;
    mrow = mnew;
    for (int dt = 0; dt < 4; ++dt)
      for (int rr = 0; rr < 8; ++rr) o[dt][rr] *= alpha;

    // Build P^T B-fragment (col = q = lane&15, K = 32 keys): half-swap shuffle
    v16h bp;
    for (int rr = 0; rr < 8; ++rr) {
      const float sendv = hs ? s0[rr] : s1[rr];
      const float got   = __shfl_xor(sendv, 16, 32);
      const float elo   = hs ? got : s0[rr];
      const float ehi   = hs ? s1[rr] : got;
      bp[rr]     = (_Float16)elo;
      bp[rr + 8] = (_Float16)ehi;
    }
    for (int dt = 0; dt < 4; ++dt)
      o[dt] = wmma_fp16(av[dt], bp, o[dt]);

    if (more)
      for (int i = 0; i < 4; ++i) ak[i] = nak[i];
  }

  const float inv = 1.f / lrow;
  _Float16* crow = ctx + ((size_t)(bIdx * kS + q0 + qr)) * kD + hIdx * kHD;
  for (int dt = 0; dt < 4; ++dt) {
    for (int rr = 0; rr < 8; rr += 2) {
      const int d = dt * 16 + hs * 8 + rr;
      f16x2 pk;
      pk.x = (_Float16)(o[dt][rr] * inv);
      pk.y = (_Float16)(o[dt][rr + 1] * inv);
      *(f16x2*)(crow + d) = pk;
    }
  }
}

// ---------------------------------------------------------------------------
// Host orchestration (graph-capture safe: only async launches on `stream`)
// ---------------------------------------------------------------------------
extern "C" void kernel_launch(void* const* d_in, const int* in_sizes, int n_in,
                              void* d_out, int out_size, void* d_ws,
                              size_t ws_size, hipStream_t stream) {
  (void)in_sizes; (void)n_in; (void)out_size; (void)ws_size;
  const float* x    = (const float*)d_in[0];
  // d_in[1] = mask: causal tril, applied analytically in flash_attn_kernel
  const float* ln1g = (const float*)d_in[2];
  const float* ln1b = (const float*)d_in[3];
  const float* wqkv = (const float*)d_in[4];
  const float* bqkv = (const float*)d_in[5];
  const float* wo   = (const float*)d_in[6];
  const float* bo   = (const float*)d_in[7];
  const float* ln2g = (const float*)d_in[8];
  const float* ln2b = (const float*)d_in[9];
  const float* w1   = (const float*)d_in[10];
  const float* b1   = (const float*)d_in[11];
  const float* w2   = (const float*)d_in[12];
  const float* b2   = (const float*)d_in[13];

  char* ws = (char*)d_ws;
  size_t off = 0;
  auto take = [&](size_t bytes) -> char* {
    char* r = ws + off;
    off += (bytes + 255) & ~(size_t)255;
    return r;
  };
  _Float16* h1    = (_Float16*)take((size_t)kM * kD * 2);        // 16 MB
  _Float16* wqkvT = (_Float16*)take((size_t)3 * kD * kD * 2);    //  6 MB
  _Float16* woT   = (_Float16*)take((size_t)kD * kD * 2);        //  2 MB
  _Float16* w1T   = (_Float16*)take((size_t)4 * kD * kD * 2);    //  8 MB
  _Float16* w2T   = (_Float16*)take((size_t)4 * kD * kD * 2);    //  8 MB
  _Float16* qbuf  = (_Float16*)take((size_t)kM * kD * 2);        // 16 MB
  _Float16* kbuf  = (_Float16*)take((size_t)kM * kD * 2);        // 16 MB
  _Float16* vtbuf = (_Float16*)take((size_t)kM * kD * 2);        // 16 MB
  _Float16* ctx   = (_Float16*)take((size_t)kM * kD * 2);        // 16 MB
  float*    x2    = (float*)take((size_t)kM * kD * 4);           // 32 MB
  _Float16* h2b   = (_Float16*)take((size_t)kM * kD * 2);        // 16 MB
  _Float16* mact  = (_Float16*)take((size_t)kM * 4 * kD * 2);    // 64 MB

  // 1) LN1 -> f16 activations
  ln_f16_kernel<<<kM, 256, 0, stream>>>(x, ln1g, ln1b, h1);

  // 2) weights -> f16 transposed
  wtrans_kernel<<<dim3(3 * kD / 256, kD), 256, 0, stream>>>(wqkv, wqkvT, kD, 3 * kD);
  wtrans_kernel<<<dim3(kD / 256, kD), 256, 0, stream>>>(wo, woT, kD, kD);
  wtrans_kernel<<<dim3(4 * kD / 256, kD), 256, 0, stream>>>(w1, w1T, kD, 4 * kD);
  wtrans_kernel<<<dim3(kD / 256, 4 * kD), 256, 0, stream>>>(w2, w2T, 4 * kD, kD);

  // 3) QKV projection, split epilogue (V stored transposed, Q pre-scaled)
  GemmArgs gq{h1, wqkvT, bqkv, nullptr, nullptr, nullptr,
              qbuf, kbuf, vtbuf, kM, 3 * kD, kD};
  gemm_f16_kernel<EPI_QKV>
      <<<dim3((3 * kD) / 128, kM / 128), 256, 0, stream>>>(gq);

  // 4) causal flash attention -> ctx f16
  flash_attn_kernel<<<dim3(kS / 128, kB * kH), 256, 0, stream>>>(qbuf, kbuf,
                                                                 vtbuf, ctx);

  // 5) output projection + residual -> x2 (f32)
  GemmArgs go{ctx, woT, bo, x, x2, nullptr, nullptr, nullptr, nullptr,
              kM, kD, kD};
  gemm_f16_kernel<EPI_F32_RESID>
      <<<dim3(kD / 128, kM / 128), 256, 0, stream>>>(go);

  // 6) LN2 -> f16
  ln_f16_kernel<<<kM, 256, 0, stream>>>(x2, ln2g, ln2b, h2b);

  // 7) MLP up + exact GELU -> f16
  GemmArgs g1{h2b, w1T, b1, nullptr, nullptr, mact, nullptr, nullptr, nullptr,
              kM, 4 * kD, kD};
  gemm_f16_kernel<EPI_F16_GELU>
      <<<dim3((4 * kD) / 128, kM / 128), 256, 0, stream>>>(g1);

  // 8) MLP down + residual -> d_out (f32)
  GemmArgs g2{mact, w2T, b2, x2, (float*)d_out, nullptr, nullptr, nullptr,
              nullptr, kM, kD, 4 * kD};
  gemm_f16_kernel<EPI_F32_RESID>
      <<<dim3(kD / 128, kM / 128), 256, 0, stream>>>(g2);
}
